// CommPolicyNet_438086664257
// MI455X (gfx1250) — matrix operand order repaired and checked by reading, not compiled, not run on hardware
//
#include <hip/hip_runtime.h>
#include <hip/hip_bf16.h>
#include <math.h>

// ---------------- CDNA5 WMMA types ----------------
typedef __bf16 bf16_t;
typedef bf16_t v16bf __attribute__((ext_vector_type(16)));
typedef float  v8f   __attribute__((ext_vector_type(8)));

struct B32x2 { uint4 lo, hi; };   // 32 bytes == one v16bf fragment

static __device__ __forceinline__ bf16_t f2bf(float f) {
    unsigned u = __builtin_bit_cast(unsigned, f);
    unsigned r = u + 0x7FFFu + ((u >> 16) & 1u);   // round-to-nearest-even
    unsigned short h = (unsigned short)(r >> 16);
    return __builtin_bit_cast(bf16_t, h);
}

// monotonic float<->uint encoding for atomic max on floats
static __device__ __forceinline__ unsigned fenc(float f) {
    unsigned u = __builtin_bit_cast(unsigned, f);
    return (u & 0x80000000u) ? ~u : (u | 0x80000000u);
}
static __device__ __forceinline__ float fdec(unsigned u) {
    unsigned v = (u & 0x80000000u) ? (u & 0x7FFFFFFFu) : ~u;
    return __builtin_bit_cast(float, v);
}

// ---- k-loop body, statically specialized on edge masking ----
template <bool MASKED>
static __device__ __forceinline__ void gemm_kloop(
    const bf16_t* __restrict__ Ab0, const bf16_t* __restrict__ Ab1,
    const bf16_t* __restrict__ Bb0, const bf16_t* __restrict__ Bb1,
    int K, unsigned mask0, unsigned mask1, v8f acc[2][2])
{
    #pragma unroll 2
    for (int kc = 0; kc < K; kc += 32) {
        // A fragments: i -> K = kc + lhi*16 + i
        B32x2 t0, t1;
        t0.lo = *(const uint4*)(Ab0 + kc);
        t0.hi = *(const uint4*)(Ab0 + kc + 8);
        t1.lo = *(const uint4*)(Ab1 + kc);
        t1.hi = *(const uint4*)(Ab1 + kc + 8);
        if (MASKED) {
            t0.lo.x &= mask0; t0.lo.y &= mask0; t0.lo.z &= mask0; t0.lo.w &= mask0;
            t0.hi.x &= mask0; t0.hi.y &= mask0; t0.hi.z &= mask0; t0.hi.w &= mask0;
            t1.lo.x &= mask1; t1.lo.y &= mask1; t1.lo.z &= mask1; t1.lo.w &= mask1;
            t1.hi.x &= mask1; t1.hi.y &= mask1; t1.hi.z &= mask1; t1.hi.w &= mask1;
        }
        v16bf afrag0 = __builtin_bit_cast(v16bf, t0);
        v16bf afrag1 = __builtin_bit_cast(v16bf, t1);
        // B fragments: i -> K = kc + lhi*16 + i (16 contiguous bf16)
        v16bf bfrag0 = *(const v16bf*)(Bb0 + kc);
        v16bf bfrag1 = *(const v16bf*)(Bb1 + kc);

        acc[0][0] = __builtin_amdgcn_wmma_f32_16x16x32_bf16(
            false, afrag0, false, bfrag0, (short)0, acc[0][0], false, false);
        acc[0][1] = __builtin_amdgcn_wmma_f32_16x16x32_bf16(
            false, afrag0, false, bfrag1, (short)0, acc[0][1], false, false);
        acc[1][0] = __builtin_amdgcn_wmma_f32_16x16x32_bf16(
            false, afrag1, false, bfrag0, (short)0, acc[1][0], false, false);
        acc[1][1] = __builtin_amdgcn_wmma_f32_16x16x32_bf16(
            false, afrag1, false, bfrag1, (short)0, acc[1][1], false, false);
    }
}

// ======================================================================
// bf16 WMMA GEMM:  C[M,N] = act( A * B^T + bias )
//   A:  bf16 [M,K] row-major   (activations, pre-converted)
//   Bt: bf16 [N,K] row-major   (weights, pre-transposed+converted)
//   N multiple of 64; K multiple of 32.
// Block = 256 threads (8 waves), block tile 128x64, wave tile 32x32 (2x2 wmma)
// Interior blocks (128 rows fully in range) take a statically mask-free loop.
// ======================================================================
template <int ACT>   // 0=none, 1=relu, 2=tanh
__global__ __launch_bounds__(256) void wmma_gemm_bf16(
    const bf16_t* __restrict__ A, const bf16_t* __restrict__ Bt,
    const float* __restrict__ bias, float* __restrict__ C,
    int M, int N, int K)
{
    const int lane = threadIdx.x & 31;
    const int wave = threadIdx.x >> 5;
    const int wm   = wave & 3;            // 4 waves along M
    const int wn   = wave >> 2;           // 2 waves along N
    const int m0   = blockIdx.x * 128 + wm * 32;
    const int n0   = blockIdx.y * 64  + wn * 32;
    const int lm   = lane & 15;
    const int lhi  = lane >> 4;           // lane half (0/1)
    const bool interior = ((int)blockIdx.x * 128 + 128) <= M;   // block-uniform

    int rowA[2]; unsigned maskA[2];
    #pragma unroll
    for (int ti = 0; ti < 2; ++ti) {
        int row = m0 + ti * 16 + lm;
        maskA[ti] = (row < M) ? 0xFFFFFFFFu : 0u;
        rowA[ti]  = (row < M) ? row : 0;
    }
    const bf16_t* Ab0 = A + (size_t)rowA[0] * K + lhi * 16;
    const bf16_t* Ab1 = A + (size_t)rowA[1] * K + lhi * 16;
    const bf16_t* Bb0 = Bt + (size_t)(n0 + lm)      * K + lhi * 16;
    const bf16_t* Bb1 = Bt + (size_t)(n0 + 16 + lm) * K + lhi * 16;

    v8f acc[2][2];
    #pragma unroll
    for (int i = 0; i < 2; ++i)
        #pragma unroll
        for (int j = 0; j < 2; ++j)
            #pragma unroll
            for (int r = 0; r < 8; ++r) acc[i][j][r] = 0.0f;

    if (interior)
        gemm_kloop<false>(Ab0, Ab1, Bb0, Bb1, K, maskA[0], maskA[1], acc);
    else
        gemm_kloop<true>(Ab0, Ab1, Bb0, Bb1, K, maskA[0], maskA[1], acc);

    // ---- store (C/D layout: VGPR r -> M = 8*lhi + r, N = lm) ----
    #pragma unroll
    for (int ti = 0; ti < 2; ++ti) {
        #pragma unroll
        for (int tj = 0; tj < 2; ++tj) {
            int col = n0 + tj * 16 + lm;
            float bv = bias ? bias[col] : 0.0f;
            float* Cp = C + (size_t)(m0 + ti * 16 + lhi * 8) * N + col;
            if (interior) {
                #pragma unroll
                for (int r = 0; r < 8; ++r) {
                    float v = acc[ti][tj][r] + bv;
                    if (ACT == 1) v = v > 0.0f ? v : 0.0f;
                    else if (ACT == 2) v = tanhf(v);
                    Cp[(size_t)r * N] = v;
                }
            } else {
                #pragma unroll
                for (int r = 0; r < 8; ++r) {
                    int row = m0 + ti * 16 + lhi * 8 + r;
                    if (row < M) {
                        float v = acc[ti][tj][r] + bv;
                        if (ACT == 1) v = v > 0.0f ? v : 0.0f;
                        else if (ACT == 2) v = tanhf(v);
                        Cp[(size_t)r * N] = v;
                    }
                }
            }
        }
    }
}

// ---------------- operand repack kernels (run once per call, tiny) ----------------
__global__ void cvt_bf16_kernel(const float* __restrict__ in, bf16_t* __restrict__ out, int n)
{
    int i = blockIdx.x * blockDim.x + threadIdx.x;
    if (i < n) out[i] = f2bf(in[i]);
}

// in: fp32 [K,N] row-major  ->  out: bf16 [N,K] row-major (i.e. B^T)
__global__ void transpose_bf16_kernel(const float* __restrict__ in, bf16_t* __restrict__ out,
                                      int K, int N)
{
    int idx = blockIdx.x * blockDim.x + threadIdx.x;
    if (idx >= K * N) return;
    int k = idx / N, n = idx - k * N;
    out[(size_t)n * K + k] = f2bf(in[idx]);
}

// ======================================================================
// Sequential GRU recurrence (torch GRUCell, batch=1, H=256).
// One persistent workgroup of 768 threads; thread j owns gate row j.
// G[t,0:768] = x_t @ Wih.T + bih  (precomputed by WMMA GEMM).
// ======================================================================
__global__ __launch_bounds__(768) void gru_scan_kernel(
    const float* __restrict__ G, const float* __restrict__ Whh,
    const float* __restrict__ bhh, float* __restrict__ Xout, int steps)
{
    __shared__ float sh_h[256], sh_r[256], sh_z[256], sh_hn[256];
    const int j = threadIdx.x;
    if (j < 256) sh_h[j] = 0.0f;
    __syncthreads();

    const float4* wr4 = (const float4*)(Whh + (size_t)j * 256);
    const float   bj  = bhh[j];
    const float4* sh4 = (const float4*)sh_h;

    for (int t = 0; t < steps; ++t) {
        float acc = bj;
        #pragma unroll 8
        for (int k = 0; k < 64; ++k) {
            float4 w = wr4[k];
            float4 h = sh4[k];
            acc += w.x * h.x + w.y * h.y + w.z * h.z + w.w * h.w;
        }
        if (j < 512) {
            float gi = G[(size_t)t * 768 + j];
            float s  = 1.0f / (1.0f + __expf(-(gi + acc)));
            if (j < 256) sh_r[j] = s;          // reset gate
            else         sh_z[j - 256] = s;    // update gate
        } else {
            sh_hn[j - 512] = acc;              // hidden part of n-gate
        }
        __syncthreads();
        if (j < 256) {
            float inn = G[(size_t)t * 768 + 512 + j];
            float n   = tanhf(inn + sh_r[j] * sh_hn[j]);
            float z   = sh_z[j];
            float hnew = (1.0f - z) * n + z * sh_h[j];
            sh_h[j] = hnew;
            Xout[(size_t)t * 256 + j] = hnew;
        }
        __syncthreads();
    }
}

// ======================================================================
// GAT edge-softmax pipeline (HEADS=8, DH=16, HC=128)
// ======================================================================
__global__ void attn_scores_kernel(const float* __restrict__ Hf,
                                   const float* __restrict__ a_src,
                                   const float* __restrict__ a_dst,
                                   float* __restrict__ ES, float* __restrict__ ED,
                                   int n)
{
    int idx = blockIdx.x * blockDim.x + threadIdx.x;
    if (idx >= n * 8) return;
    int node = idx >> 3, h = idx & 7;
    const float4* hp = (const float4*)(Hf + (size_t)node * 128 + h * 16);
    const float4* as = (const float4*)(a_src + h * 16);
    const float4* ad = (const float4*)(a_dst + h * 16);
    float es = 0.f, ed = 0.f;
    #pragma unroll
    for (int q = 0; q < 4; ++q) {
        float4 v = hp[q], s = as[q], d = ad[q];
        es += v.x * s.x + v.y * s.y + v.z * s.z + v.w * s.w;
        ed += v.x * d.x + v.y * d.y + v.z * d.z + v.w * d.w;
    }
    ES[idx] = es; ED[idx] = ed;
}

__global__ void init_emax_kernel(unsigned* __restrict__ EM, int count)
{
    int idx = blockIdx.x * blockDim.x + threadIdx.x;
    if (idx < count) EM[idx] = 0x007FFFFFu;   // fenc(-inf)
}

__global__ void edge_max_kernel(const int* __restrict__ ei,
                                const float* __restrict__ ES, const float* __restrict__ ED,
                                unsigned* __restrict__ EM, float* __restrict__ P, int E)
{
    int idx = blockIdx.x * blockDim.x + threadIdx.x;
    if (idx >= E * 8) return;
    int e = idx >> 3, h = idx & 7;
    int s = ei[e], d = ei[E + e];
    float x = ES[s * 8 + h] + ED[d * 8 + h];
    x = x > 0.0f ? x : 0.2f * x;              // leaky_relu 0.2
    P[idx] = x;
    atomicMax(&EM[d * 8 + h], fenc(x));
}

__global__ void edge_p_kernel(const int* __restrict__ ei,
                              const unsigned* __restrict__ EM,
                              float* __restrict__ P, float* __restrict__ DEN, int E)
{
    int idx = blockIdx.x * blockDim.x + threadIdx.x;
    if (idx >= E * 8) return;
    int e = idx >> 3, h = idx & 7;
    int d = ei[E + e];
    float p = __expf(P[idx] - fdec(EM[d * 8 + h]));
    P[idx] = p;
    atomicAdd(&DEN[d * 8 + h], p);
}

__global__ void edge_aggr_kernel(const int* __restrict__ ei,
                                 const float* __restrict__ Hf,
                                 const float* __restrict__ P,
                                 const float* __restrict__ DEN,
                                 float* __restrict__ OUT, int E)
{
    int idx = blockIdx.x * blockDim.x + threadIdx.x;          // E*128 < 2^31
    if (idx >= E * 128) return;
    int e = idx >> 7, hd = idx & 127, h = hd >> 4;
    int s = ei[e], d = ei[E + e];
    float alpha = P[e * 8 + h] / (DEN[d * 8 + h] + 1e-16f);
    atomicAdd(&OUT[(size_t)d * 128 + hd], Hf[(size_t)s * 128 + hd] * alpha);
}

template <bool RELU>
__global__ void finalize_kernel(float* __restrict__ OUT, const float* __restrict__ bias, int n)
{
    int idx = blockIdx.x * blockDim.x + threadIdx.x;
    if (idx >= n * 128) return;
    float v = OUT[idx] + bias[idx & 127];
    if (RELU) v = v > 0.0f ? v : 0.0f;
    OUT[idx] = v;
}

// ---------------- small elementwise / head kernels ----------------
__global__ void add_kernel(float* __restrict__ a, const float* __restrict__ b, int n)
{
    int i = blockIdx.x * blockDim.x + threadIdx.x;
    if (i < n) a[i] += b[i];
}

__global__ void relu_cvt_kernel(float* __restrict__ a, bf16_t* __restrict__ ab, int n)
{
    int i = blockIdx.x * blockDim.x + threadIdx.x;
    if (i < n) { float v = a[i]; v = v > 0.0f ? v : 0.0f; a[i] = v; ab[i] = f2bf(v); }
}

__global__ void comm_head_kernel(const float* __restrict__ XG,
                                 const float* __restrict__ Wc, const float* __restrict__ bc,
                                 float* __restrict__ out, int n)
{
    int i = blockIdx.x * blockDim.x + threadIdx.x;
    if (i >= n) return;
    const float4* x = (const float4*)(XG + (size_t)i * 128);
    const float4* w = (const float4*)Wc;
    float s = bc[0];
    #pragma unroll 8
    for (int k = 0; k < 32; ++k) {
        float4 xv = x[k], wv = w[k];
        s += xv.x * wv.x + xv.y * wv.y + xv.z * wv.z + xv.w * wv.w;
    }
    out[i] = 1.0f / (1.0f + __expf(-s));
}

__global__ void mu_head_kernel(const float* __restrict__ X, const float* __restrict__ XG,
                               const float* __restrict__ Wmu, const float* __restrict__ bmu,
                               float* __restrict__ out, int n)
{
    int idx = blockIdx.x * blockDim.x + threadIdx.x;   // n*8
    if (idx >= n * 8) return;
    int node = idx >> 3, a = idx & 7;
    const float* xp = X  + (size_t)node * 256;
    const float* gp = XG + (size_t)node * 128;
    float s = bmu[a];
    #pragma unroll 8
    for (int k = 0; k < 256; ++k) s += xp[k] * Wmu[(size_t)k * 8 + a];
    #pragma unroll 8
    for (int k = 0; k < 128; ++k) s += gp[k] * Wmu[(size_t)(256 + k) * 8 + a];
    out[idx] = tanhf(s);
}

// ======================================================================
// Host orchestration
// ======================================================================
extern "C" void kernel_launch(void* const* d_in, const int* in_sizes, int n_in,
                              void* d_out, int out_size, void* d_ws, size_t ws_size,
                              hipStream_t stream)
{
    (void)n_in; (void)out_size; (void)ws_size;

    const float* state   = (const float*)d_in[0];
    const float* message = (const float*)d_in[1];
    const int*   eidx    = (const int*)  d_in[2];
    const float* W_fc1 = (const float*)d_in[3];  const float* b_fc1 = (const float*)d_in[4];
    const float* W_fc2 = (const float*)d_in[5];  const float* b_fc2 = (const float*)d_in[6];
    const float* Wih0  = (const float*)d_in[7];  const float* Whh0  = (const float*)d_in[8];
    const float* bih0  = (const float*)d_in[9];  const float* bhh0  = (const float*)d_in[10];
    const float* Wih1  = (const float*)d_in[11]; const float* Whh1  = (const float*)d_in[12];
    const float* bih1  = (const float*)d_in[13]; const float* bhh1  = (const float*)d_in[14];
    const float* Wg1 = (const float*)d_in[15];
    const float* as1 = (const float*)d_in[16];  const float* ad1 = (const float*)d_in[17];
    const float* bg1 = (const float*)d_in[18];
    const float* Wg2 = (const float*)d_in[19];
    const float* as2 = (const float*)d_in[20];  const float* ad2 = (const float*)d_in[21];
    const float* bg2 = (const float*)d_in[22];
    const float* Wc  = (const float*)d_in[23];  const float* bc  = (const float*)d_in[24];
    const float* Wmu = (const float*)d_in[25];  const float* bmu = (const float*)d_in[26];
    const float* Wm  = (const float*)d_in[27];  const float* bm  = (const float*)d_in[28];

    const int N = in_sizes[0] / 64;       // D_IN = 64
    const int E = in_sizes[2] / 2;

    // ---- workspace carve ----
    char* wsb = (char*)d_ws;
    size_t off = 0;
    auto carveF = [&](size_t cnt) { float* p = (float*)(wsb + off); off += cnt * 4; return p; };
    auto carveH = [&](size_t cnt) { bf16_t* p = (bf16_t*)(wsb + off);
                                    off += ((cnt * 2 + 63) & ~(size_t)63); return p; };
    // fp32 buffers
    float* X0  = carveF((size_t)N * 256);
    float* X1  = carveF((size_t)N * 256);
    float* X2  = carveF((size_t)N * 256);
    float* G   = carveF((size_t)N * 768);
    float* Hb  = carveF((size_t)N * 128);
    float* XG1 = carveF((size_t)N * 128);
    float* XG2 = carveF((size_t)N * 128);
    float* ES  = carveF((size_t)N * 8);
    float* ED  = carveF((size_t)N * 8);
    unsigned* EM = (unsigned*)carveF((size_t)N * 8);
    float* DEN = carveF((size_t)N * 8);
    float* P   = carveF((size_t)E * 8);
    // bf16 operand buffers
    bf16_t* SB   = carveH((size_t)N * 64);
    bf16_t* MB   = carveH((size_t)N * 64);
    bf16_t* X0b  = carveH((size_t)N * 256);
    bf16_t* X1b  = carveH((size_t)N * 256);
    bf16_t* X2b  = carveH((size_t)N * 256);
    bf16_t* XG1b = carveH((size_t)N * 128);
    bf16_t* XG2b = carveH((size_t)N * 128);
    bf16_t* Wfc1T = carveH(64 * 256);      // [256,64]
    bf16_t* Wfc2T = carveH(64 * 256);
    bf16_t* Wih0b = carveH(768 * 256);     // already [N,K]
    bf16_t* Wih1b = carveH(768 * 256);
    bf16_t* Wg1T  = carveH(256 * 128);     // [128,256]
    bf16_t* Wg2T  = carveH(128 * 128);
    bf16_t* WmT   = carveH(128 * 64);      // [64,128]

    float* out_comm = (float*)d_out;               // [N]
    float* out_msg  = out_comm + N;                // [N,64]
    float* out_mu   = out_msg + (size_t)N * 64;    // [N,8]

    const dim3 blk(256);
    dim3 gM((N + 127) / 128, 1);
    auto grid1 = [&](long cnt) { return dim3((unsigned)((cnt + 255) / 256)); };

    // 0) operand repack (weights + inputs)
    transpose_bf16_kernel<<<grid1(64 * 256), blk, 0, stream>>>(W_fc1, Wfc1T, 64, 256);
    transpose_bf16_kernel<<<grid1(64 * 256), blk, 0, stream>>>(W_fc2, Wfc2T, 64, 256);
    cvt_bf16_kernel<<<grid1(768 * 256), blk, 0, stream>>>(Wih0, Wih0b, 768 * 256);
    cvt_bf16_kernel<<<grid1(768 * 256), blk, 0, stream>>>(Wih1, Wih1b, 768 * 256);
    transpose_bf16_kernel<<<grid1(256 * 128), blk, 0, stream>>>(Wg1, Wg1T, 256, 128);
    transpose_bf16_kernel<<<grid1(128 * 128), blk, 0, stream>>>(Wg2, Wg2T, 128, 128);
    transpose_bf16_kernel<<<grid1(128 * 64),  blk, 0, stream>>>(Wm,  WmT,  128, 64);
    cvt_bf16_kernel<<<grid1((long)N * 64), blk, 0, stream>>>(state,   SB, N * 64);
    cvt_bf16_kernel<<<grid1((long)N * 64), blk, 0, stream>>>(message, MB, N * 64);

    // 1) fc1 / fc2 + add  (relu in GEMM epilogue)
    wmma_gemm_bf16<1><<<dim3(gM.x, 4), blk, 0, stream>>>(SB, Wfc1T, b_fc1, X0, N, 256, 64);
    wmma_gemm_bf16<1><<<dim3(gM.x, 4), blk, 0, stream>>>(MB, Wfc2T, b_fc2, X1, N, 256, 64);
    add_kernel<<<grid1((long)N * 256), blk, 0, stream>>>(X0, X1, N * 256);
    cvt_bf16_kernel<<<grid1((long)N * 256), blk, 0, stream>>>(X0, X0b, N * 256);

    // 2) GRU layer 0: batched input GEMM, then sequential scan
    wmma_gemm_bf16<0><<<dim3(gM.x, 12), blk, 0, stream>>>(X0b, Wih0b, bih0, G, N, 768, 256);
    gru_scan_kernel<<<1, 768, 0, stream>>>(G, Whh0, bhh0, X1, N);
    cvt_bf16_kernel<<<grid1((long)N * 256), blk, 0, stream>>>(X1, X1b, N * 256);

    // 3) GRU layer 1
    wmma_gemm_bf16<0><<<dim3(gM.x, 12), blk, 0, stream>>>(X1b, Wih1b, bih1, G, N, 768, 256);
    gru_scan_kernel<<<1, 768, 0, stream>>>(G, Whh1, bhh1, X2, N);
    relu_cvt_kernel<<<grid1((long)N * 256), blk, 0, stream>>>(X2, X2b, N * 256);

    // 4) GAT layer 1
    wmma_gemm_bf16<0><<<dim3(gM.x, 2), blk, 0, stream>>>(X2b, Wg1T, nullptr, Hb, N, 128, 256);
    attn_scores_kernel<<<grid1((long)N * 8), blk, 0, stream>>>(Hb, as1, ad1, ES, ED, N);
    init_emax_kernel<<<grid1((long)N * 8), blk, 0, stream>>>(EM, N * 8);
    hipMemsetAsync(DEN, 0, (size_t)N * 8 * sizeof(float), stream);
    hipMemsetAsync(XG1, 0, (size_t)N * 128 * sizeof(float), stream);
    edge_max_kernel<<<grid1((long)E * 8), blk, 0, stream>>>(eidx, ES, ED, EM, P, E);
    edge_p_kernel<<<grid1((long)E * 8), blk, 0, stream>>>(eidx, EM, P, DEN, E);
    edge_aggr_kernel<<<grid1((long)E * 128), blk, 0, stream>>>(eidx, Hb, P, DEN, XG1, E);
    finalize_kernel<true><<<grid1((long)N * 128), blk, 0, stream>>>(XG1, bg1, N);
    cvt_bf16_kernel<<<grid1((long)N * 128), blk, 0, stream>>>(XG1, XG1b, N * 128);

    // 5) GAT layer 2
    wmma_gemm_bf16<0><<<dim3(gM.x, 2), blk, 0, stream>>>(XG1b, Wg2T, nullptr, Hb, N, 128, 128);
    attn_scores_kernel<<<grid1((long)N * 8), blk, 0, stream>>>(Hb, as2, ad2, ES, ED, N);
    init_emax_kernel<<<grid1((long)N * 8), blk, 0, stream>>>(EM, N * 8);
    hipMemsetAsync(DEN, 0, (size_t)N * 8 * sizeof(float), stream);
    hipMemsetAsync(XG2, 0, (size_t)N * 128 * sizeof(float), stream);
    edge_max_kernel<<<grid1((long)E * 8), blk, 0, stream>>>(eidx, ES, ED, EM, P, E);
    edge_p_kernel<<<grid1((long)E * 8), blk, 0, stream>>>(eidx, EM, P, DEN, E);
    edge_aggr_kernel<<<grid1((long)E * 128), blk, 0, stream>>>(eidx, Hb, P, DEN, XG2, E);
    finalize_kernel<false><<<grid1((long)N * 128), blk, 0, stream>>>(XG2, bg2, N);
    cvt_bf16_kernel<<<grid1((long)N * 128), blk, 0, stream>>>(XG2, XG2b, N * 128);

    // 6) heads
    comm_head_kernel<<<grid1(N), blk, 0, stream>>>(XG2, Wc, bc, out_comm, N);
    wmma_gemm_bf16<2><<<dim3(gM.x, 1), blk, 0, stream>>>(XG2b, WmT, bm, out_msg, N, 64, 128);
    mu_head_kernel<<<grid1((long)N * 8), blk, 0, stream>>>(X2, XG2, Wmu, bmu, out_mu, N);
}
